// NTN_50560355008672
// MI455X (gfx1250) — matrix operand order repaired and checked by reading, not compiled, not run on hardware
//
#include <hip/hip_runtime.h>
#include <hip/hip_bf16.h>

typedef __attribute__((ext_vector_type(16))) __bf16 v16bf;
typedef __attribute__((ext_vector_type(8)))  float  v8f;

#define B_TOT   2048
#define D_DIM   512
#define K_DIM   32
#define F_MAIN  (D_DIM * D_DIM)      // 262144 (d,e) contraction rows
#define F_EXT   (2 * D_DIM)          // 1024 appended rows: concat(x1,x2) x V^T
#define F_TOT   (F_MAIN + F_EXT)     // 263168
#define SPLIT   16                   // contraction split across workgroups
#define MT      4                    // M-tiles (of 16 batches) per workgroup
#define WGB     (MT * 16)            // 64 batches per workgroup
#define ROWS    1032                 // cat-tile row stride (bf16 units): 1024 + 8 pad
#define HSTR    33                   // h-tile row stride (floats)

#define PKMUL(dst, s, w) \
  asm("v_pk_mul_bf16 %0, %1, %2" : "=v"(dst) : "v"(s), "v"(w))

// ---------------------------------------------------------------------------
// Kernel 1: pack W [512,512,32] and V^T [1024,32] to bf16 k-major 32f x 32k blocks in ws;
//           also zero the cross-split h accumulation buffer. 8 elems/thread.
// ---------------------------------------------------------------------------
__global__ void ntn_pack(const float* __restrict__ W, const float* __restrict__ V,
                         uint4* __restrict__ wsp, float* __restrict__ hbuf) {
  long p = (long)blockIdx.x * blockDim.x + threadIdx.x;   // one uint4 = 8 bf16
  const long NW = (long)F_MAIN * K_DIM;                   // 8,388,608
  const long NT = (long)F_TOT * K_DIM;                    // 8,421,376
  if (p < (long)B_TOT * K_DIM) hbuf[p] = 0.0f;            // zero h accumulator
  long e0 = p * 8;
  if (e0 >= NT) return;
  float f[8];
  {                                 // 32f x 32k blocks stored k-major: [k][f]
    long blk = e0 >> 10, o = e0 & 1023;
    long kk = o >> 5, fr = blk * 32 + (o & 31);
#pragma unroll
    for (int i = 0; i < 8; ++i)
      f[i] = (fr < F_MAIN) ? W[(fr + i) * K_DIM + kk] : V[kk * (2 * D_DIM) + (fr + i - F_MAIN)];
  }
  union { __bf16 h[8]; uint4 q; } pk;
#pragma unroll
  for (int i = 0; i < 8; ++i) pk.h[i] = (__bf16)f[i];
  wsp[p] = pk.q;
}

// ---------------------------------------------------------------------------
// Kernel 2: main fused GEMM. Grid = 32 batch-groups x SPLIT. 512 threads.
// Each wave: 4 M-tiles x 2 k-tiles of accumulators over a 2-d contraction slice.
// ---------------------------------------------------------------------------
__global__ __launch_bounds__(512)
void ntn_main(const float* __restrict__ x1, const float* __restrict__ x2,
              const __bf16* __restrict__ wsW, float* __restrict__ hbuf) {
  __shared__ __bf16 cat[WGB * ROWS];     // per-batch row: [x1(512) | x2(512) | pad]
  __shared__ float  htile[WGB * HSTR];   // workgroup h accumulation tile

  const int t     = threadIdx.x;
  const int wave  = t >> 5;
  const int lane  = t & 31;
  const int bgrp  = blockIdx.x / SPLIT;
  const int split = blockIdx.x % SPLIT;
  const int bbase = bgrp * WGB;

  // Zero h-tile.
  for (int idx = t; idx < WGB * HSTR; idx += 512) htile[idx] = 0.0f;

  // Stage x1/x2 tile -> bf16 cat tile (float4 in, b64 out, coalesced).
  for (int u = t; u < WGB * 128; u += 512) {
    const int r = u >> 7, c4 = u & 127;
    float4 a = *(const float4*)(x1 + (bbase + r) * D_DIM + c4 * 4);
    float4 b = *(const float4*)(x2 + (bbase + r) * D_DIM + c4 * 4);
    union { __bf16 h[4]; uint2 u2; } pa, pb;
    pa.h[0] = (__bf16)a.x; pa.h[1] = (__bf16)a.y; pa.h[2] = (__bf16)a.z; pa.h[3] = (__bf16)a.w;
    pb.h[0] = (__bf16)b.x; pb.h[1] = (__bf16)b.y; pb.h[2] = (__bf16)b.z; pb.h[3] = (__bf16)b.w;
    *(uint2*)(cat + r * ROWS + c4 * 4)       = pa.u2;
    *(uint2*)(cat + r * ROWS + 512 + c4 * 4) = pb.u2;
  }
  __syncthreads();

  const int hlf  = lane >> 4;   // A/C half select
  const int mrow = lane & 15;   // batch row within an M-tile this lane feeds

  v8f acc[MT][2];
#pragma unroll
  for (int m = 0; m < MT; ++m) { acc[m][0] = {}; acc[m][1] = {}; }

  // Preload the 8 x1 scalars (4 M-tiles x 2 d-values) as packed bf16 pairs.
  const int dbase = split * (D_DIM / SPLIT) + wave * 2;   // 2 d-values per wave
  unsigned ssv[MT][2];
#pragma unroll
  for (int m = 0; m < MT; ++m)
#pragma unroll
    for (int dd = 0; dd < 2; ++dd) {
      unsigned short sx = __builtin_bit_cast(
          unsigned short, cat[(m * 16 + mrow) * ROWS + dbase + dd]);
      ssv[m][dd] = (unsigned)sx | ((unsigned)sx << 16);
    }

  const unsigned fbase = (unsigned)dbase * (unsigned)D_DIM;

  // ---- main region: e-outer so x2 windows are reused across d and k-tiles ----
  for (int e0 = 0; e0 < D_DIM; e0 += 32) {
    uint4 wnd[MT][2];
#pragma unroll
    for (int m = 0; m < MT; ++m) {
      const __bf16* wp = cat + (m * 16 + mrow) * ROWS + 512 + e0 + hlf * 8;
      wnd[m][0] = *(const uint4*)(wp);
      wnd[m][1] = *(const uint4*)(wp + 16);
    }
#pragma unroll
    for (int dd = 0; dd < 2; ++dd) {
      const char* bp = (const char*)wsW +
          (((fbase + (unsigned)dd * (unsigned)D_DIM + (unsigned)e0) + (unsigned)mrow) << 6) + hlf * 16;
      union { uint4 q[2]; v16bf v; } B0, B1;
      B0.q[0] = *(const uint4*)(bp);
      B0.q[1] = *(const uint4*)(bp + 32);
      B1.q[0] = *(const uint4*)(bp + 1024);
      B1.q[1] = *(const uint4*)(bp + 1056);
#pragma unroll
      for (int m = 0; m < MT; ++m) {
        union { unsigned u[8]; v16bf v; } A;
        PKMUL(A.u[0], ssv[m][dd], wnd[m][0].x);
        PKMUL(A.u[1], ssv[m][dd], wnd[m][0].y);
        PKMUL(A.u[2], ssv[m][dd], wnd[m][0].z);
        PKMUL(A.u[3], ssv[m][dd], wnd[m][0].w);
        PKMUL(A.u[4], ssv[m][dd], wnd[m][1].x);
        PKMUL(A.u[5], ssv[m][dd], wnd[m][1].y);
        PKMUL(A.u[6], ssv[m][dd], wnd[m][1].z);
        PKMUL(A.u[7], ssv[m][dd], wnd[m][1].w);
        acc[m][0] = __builtin_amdgcn_wmma_f32_16x16x32_bf16(
            false, A.v, false, B0.v, (short)0, acc[m][0], false, false);
        acc[m][1] = __builtin_amdgcn_wmma_f32_16x16x32_bf16(
            false, A.v, false, B1.v, (short)0, acc[m][1], false, false);
      }
    }
  }

  // ---- extension region (split 0 only): folds V * concat(x1,x2) into acc ----
  if (split == 0) {
#pragma unroll
    for (int c = 0; c < 2; ++c) {
      const int j0 = wave * 64 + c * 32;
      const char* bp = (const char*)wsW +
                       (((unsigned)(F_MAIN + j0) + (unsigned)mrow) << 6) + hlf * 16;
      union { uint4 q[2]; v16bf v; } B0, B1;
      B0.q[0] = *(const uint4*)(bp);
      B0.q[1] = *(const uint4*)(bp + 32);
      B1.q[0] = *(const uint4*)(bp + 1024);
      B1.q[1] = *(const uint4*)(bp + 1056);
#pragma unroll
      for (int m = 0; m < MT; ++m) {
        const __bf16* ap = cat + (m * 16 + mrow) * ROWS + j0 + hlf * 8;
        union { uint4 q[2]; v16bf v; } A;
        A.q[0] = *(const uint4*)(ap);
        A.q[1] = *(const uint4*)(ap + 16);
        acc[m][0] = __builtin_amdgcn_wmma_f32_16x16x32_bf16(
            false, A.v, false, B0.v, (short)0, acc[m][0], false, false);
        acc[m][1] = __builtin_amdgcn_wmma_f32_16x16x32_bf16(
            false, A.v, false, B1.v, (short)0, acc[m][1], false, false);
      }
    }
  }

  // ---- reduce 16 waves via LDS float atomics, then combine splits globally ----
#pragma unroll
  for (int m = 0; m < MT; ++m)
#pragma unroll
    for (int i = 0; i < 8; ++i) {
      const int row = m * 16 + i + 8 * hlf;   // C layout: VGPR i -> M = i + 8*half
      atomicAdd(&htile[row * HSTR + mrow],      acc[m][0][i]);
      atomicAdd(&htile[row * HSTR + 16 + mrow], acc[m][1][i]);
    }
  __syncthreads();
#pragma unroll
  for (int p = 0; p < (WGB * K_DIM) / 512; ++p) {
    const int idx = t + p * 512;
    const int b = idx >> 5, k = idx & 31;
    atomicAdd(&hbuf[(bbase + b) * K_DIM + k], htile[b * HSTR + k]);
  }
}

// ---------------------------------------------------------------------------
// Kernel 3: out[b] = sum_k U[k] * relu(h[b,k] + bias[k])
// ---------------------------------------------------------------------------
__global__ void ntn_epilogue(const float* __restrict__ hbuf,
                             const float* __restrict__ U,
                             const float* __restrict__ bias,
                             float* __restrict__ out) {
  const int t = threadIdx.x;
  const int b = blockIdx.x * 8 + (t >> 5);   // one wave per batch element
  const int k = t & 31;
  float val = hbuf[b * K_DIM + k] + bias[k];
  val = fmaxf(val, 0.0f) * U[k];
  for (int o = 16; o > 0; o >>= 1) val += __shfl_xor(val, o, 32);
  if (k == 0) out[b] = val;
}

// ---------------------------------------------------------------------------
extern "C" void kernel_launch(void* const* d_in, const int* in_sizes, int n_in,
                              void* d_out, int out_size, void* d_ws, size_t ws_size,
                              hipStream_t stream) {
  (void)in_sizes; (void)n_in; (void)out_size; (void)ws_size;
  const float* x1  = (const float*)d_in[0];
  const float* x2  = (const float*)d_in[1];
  const float* W   = (const float*)d_in[2];
  const float* V   = (const float*)d_in[3];
  const float* U   = (const float*)d_in[4];
  const float* bia = (const float*)d_in[5];
  float* out = (float*)d_out;

  // ws layout: [0, F_TOT*32*2) bf16 packed W|V^T, then float h[2048][32]
  __bf16* wsW  = (__bf16*)d_ws;
  float*  hbuf = (float*)((char*)d_ws + (size_t)F_TOT * K_DIM * 2);

  const long nq  = (long)F_TOT * K_DIM / 8;                // uint4 units
  const int  tpb = 256;
  ntn_pack<<<(unsigned)((nq + tpb - 1) / tpb), tpb, 0, stream>>>(
      W, V, (uint4*)wsW, hbuf);

  ntn_main<<<(B_TOT / WGB) * SPLIT, 512, 0, stream>>>(x1, x2, wsW, hbuf);

  ntn_epilogue<<<B_TOT / 8, 256, 0, stream>>>(hbuf, U, bia, out);
}
